// FastRCNNPredictorPairedSortedAtten_14542759264791
// MI455X (gfx1250) — matrix operand order repaired and checked by reading, not compiled, not run on hardware
//
#include <hip/hip_runtime.h>

typedef _Float16 half_t;
typedef _Float16 v16h __attribute__((ext_vector_type(16)));
typedef _Float16 h4   __attribute__((ext_vector_type(4)));
typedef float    v8f  __attribute__((ext_vector_type(8)));
typedef int      v4i_vs __attribute__((vector_size(16)));   // matches builtin param type

#define BM 128
#define BN 128
#define BKT 32
#define LDSS 40   // padded LDS row stride in halves (80B): kills bank conflicts, keeps 16B align

union Frag { v16h v; unsigned u[8]; };

// ---- CDNA5 async global->LDS copy (ASYNCcnt path), 16 bytes per lane ----
// clang-22 signature (revealed by diagnostics):
//   void __builtin_amdgcn_global_load_async_to_lds_b128(
//       v4i addrspace(1)* gsrc, v4i addrspace(3)* lds, imm int offset, imm int cpol)
#if __has_builtin(__builtin_amdgcn_global_load_async_to_lds_b128)
#define ASYNC_LDS_CP16(ldsdst, gsrc)                                              \
  __builtin_amdgcn_global_load_async_to_lds_b128(                                 \
      (__attribute__((address_space(1))) v4i_vs*)(void*)(gsrc),                   \
      (__attribute__((address_space(3))) v4i_vs*)(void*)(ldsdst), 0, 0)
#else
#define ASYNC_LDS_CP16(ldsdst, gsrc)                                              \
  do {                                                                            \
    unsigned _lo = (unsigned)(unsigned long long)(void*)(ldsdst);                 \
    asm volatile("global_load_async_to_lds_b128 %0, %1, off"                      \
                 :: "v"(_lo), "v"((unsigned long long)(const void*)(gsrc))        \
                 : "memory");                                                     \
  } while (0)
#endif

__device__ __forceinline__ void async_wait0() {
#if __has_builtin(__builtin_amdgcn_s_wait_asynccnt)
  __builtin_amdgcn_s_wait_asynccnt(0);
#else
  asm volatile("s_wait_asynccnt 0x0" ::: "memory");
#endif
}

// ---------------------------------------------------------------------------
// Generic batched GEMM:  C[m,n] = alpha * sum_k Ae[m,k]*Be[n,k] + bias[n] + Res[m,n]
//   Ae[m,k] = aColMajor ? A[k*lda+m] : A[m*lda+k]
//   Be[n,k] = bKN       ? B[k*ldb+n] : B[n*ldb+k]
// f16 operands, f32 accumulate via v_wmma_f32_16x16x32_f16.
// ---------------------------------------------------------------------------
__global__ __launch_bounds__(256)
void gemm_f16_wmma(const half_t* __restrict__ A, long long strideA, int lda, int aColMajor,
                   const half_t* __restrict__ B, long long strideB, int ldb, int bKN,
                   const float* __restrict__ bias,
                   const float* __restrict__ Res, long long strideRes, int ldr,
                   float alpha,
                   float* __restrict__ Cf, long long strideCf, int ldcf,
                   half_t* __restrict__ Ch, long long strideCh, int ldch,
                   int M, int N, int K)
{
  __shared__ half_t As[2][BM][LDSS];
  __shared__ half_t Bs[2][BN][LDSS];

  const int bz = blockIdx.z;
  A += (long long)bz * strideA;
  B += (long long)bz * strideB;
  if (Res) Res += (long long)bz * strideRes;
  if (Cf)  Cf  += (long long)bz * strideCf;
  if (Ch)  Ch  += (long long)bz * strideCh;

  const int m0 = blockIdx.y * BM;
  const int n0 = blockIdx.x * BN;
  const int tid   = threadIdx.x;
  const int lane  = tid & 31;
  const int wid   = tid >> 5;
  const int waveM = wid & 3;   // 0..3 -> 32-row strip
  const int waveN = wid >> 2;  // 0..1 -> 64-col strip
  const int lmod  = lane & 15;
  const int lhi   = lane >> 4; // 0 or 1

  const bool aVec   = (!aColMajor) && ((lda & 3) == 0);
  const bool bVec   = ((ldb & 3) == 0);
  const bool aAsync = (!aColMajor) && ((lda & 7) == 0) && (m0 + BM <= M);
  const bool bAsync = (!bKN)       && ((ldb & 7) == 0) && (n0 + BN <= N);

  auto stageA = [&](int kt, int buf) {
    const int k0 = kt * BKT;
    if (aAsync && (k0 + BKT <= K)) {
      // async: 128 rows x 64B = 512 x 16B chunks, 2 per thread, no VGPR round-trip
      #pragma unroll
      for (int it = 0; it < 2; ++it) {
        int q  = tid + it * 256;       // 0..511
        int r  = q >> 2;               // row
        int c8 = (q & 3) << 3;         // half offset {0,8,16,24}
        ASYNC_LDS_CP16(&As[buf][r][c8], A + (long long)(m0 + r) * lda + (k0 + c8));
      }
      return;
    }
    if (!aColMajor) {
      #pragma unroll
      for (int it = 0; it < 4; ++it) {
        int q  = tid + it * 256;      // 0..1023 quads
        int r  = q >> 3;              // row in tile
        int c4 = (q & 7) << 2;        // k offset
        int gm = m0 + r, gk = k0 + c4;
        h4 val; val[0] = val[1] = val[2] = val[3] = (half_t)0.f;
        if (gm < M) {
          if (aVec && gk + 3 < K) {
            val = *(const h4*)(A + (long long)gm * lda + gk);
          } else {
            #pragma unroll
            for (int j = 0; j < 4; ++j)
              if (gk + j < K) val[j] = A[(long long)gm * lda + gk + j];
          }
        }
        *(h4*)&As[buf][r][c4] = val;
      }
    } else {
      #pragma unroll
      for (int it = 0; it < 16; ++it) {
        int e = tid + it * 256;       // 0..4095 elements
        int r = e >> 5;
        int c = e & 31;
        int gm = m0 + r, gk = k0 + c;
        half_t v = (half_t)0.f;
        if (gm < M && gk < K) v = A[(long long)gk * lda + gm];
        As[buf][r][c] = v;
      }
    }
  };

  auto stageB = [&](int kt, int buf) {
    const int k0 = kt * BKT;
    if (bAsync && (k0 + BKT <= K)) {
      #pragma unroll
      for (int it = 0; it < 2; ++it) {
        int q  = tid + it * 256;
        int r  = q >> 2;
        int c8 = (q & 3) << 3;
        ASYNC_LDS_CP16(&Bs[buf][r][c8], B + (long long)(n0 + r) * ldb + (k0 + c8));
      }
      return;
    }
    if (!bKN) {
      #pragma unroll
      for (int it = 0; it < 4; ++it) {
        int q  = tid + it * 256;
        int r  = q >> 3;              // n in tile
        int c4 = (q & 7) << 2;        // k
        int gn = n0 + r, gk = k0 + c4;
        h4 val; val[0] = val[1] = val[2] = val[3] = (half_t)0.f;
        if (gn < N) {
          if (bVec && gk + 3 < K) {
            val = *(const h4*)(B + (long long)gn * ldb + gk);
          } else {
            #pragma unroll
            for (int j = 0; j < 4; ++j)
              if (gk + j < K) val[j] = B[(long long)gn * ldb + gk + j];
          }
        }
        *(h4*)&Bs[buf][r][c4] = val;
      }
    } else {
      // source is (K,N) row-major: vectorize along contiguous n, transpose into LDS
      #pragma unroll
      for (int it = 0; it < 4; ++it) {
        int q  = tid + it * 256;      // 0..1023
        int kk = q >> 5;              // 0..31
        int n4 = (q & 31) << 2;       // 0..124
        int gk = k0 + kk, gn = n0 + n4;
        h4 val; val[0] = val[1] = val[2] = val[3] = (half_t)0.f;
        if (gk < K) {
          if (bVec && gn + 3 < N) {
            val = *(const h4*)(B + (long long)gk * ldb + gn);
          } else {
            #pragma unroll
            for (int j = 0; j < 4; ++j)
              if (gn + j < N) val[j] = B[(long long)gk * ldb + gn + j];
          }
        }
        #pragma unroll
        for (int j = 0; j < 4; ++j) Bs[buf][n4 + j][kk] = val[j];
      }
    }
  };

  v8f acc[2][4];
  const v8f zero = {0.f,0.f,0.f,0.f,0.f,0.f,0.f,0.f};
  #pragma unroll
  for (int mt = 0; mt < 2; ++mt)
    #pragma unroll
    for (int nt = 0; nt < 4; ++nt) acc[mt][nt] = zero;

  const int nk = (K + BKT - 1) / BKT;
  stageA(0, 0); stageB(0, 0);
  async_wait0();
  __syncthreads();

  for (int kt = 0; kt < nk; ++kt) {
    const int cur = kt & 1;
    if (kt + 1 < nk) { stageA(kt + 1, cur ^ 1); stageB(kt + 1, cur ^ 1); }

    Frag af[2], bf[4];
    // A fragment: 16-bit A 16x32 layout (lanes 0-15 K:{0..7,16..23}, lanes 16-31 K:{8..15,24..31})
    #pragma unroll
    for (int mt = 0; mt < 2; ++mt) {
      const int row = waveM * 32 + mt * 16 + lmod;
      const int khi = lhi * 8;
      #pragma unroll
      for (int p = 0; p < 8; ++p) {
        const int kk = (p < 4 ? 2 * p : 16 + 2 * (p - 4)) + khi;
        af[mt].u[p] = *(const unsigned*)&As[cur][row][kk];
      }
    }
    // B fragment: lanes 0-15 K:{0..7,16..23}, lanes 16-31 K:{8..15,24..31}, n = lane%16
    #pragma unroll
    for (int nt = 0; nt < 4; ++nt) {
      const int col = waveN * 64 + nt * 16 + lmod;
      const int kof = lhi * 8;
      #pragma unroll
      for (int p = 0; p < 8; ++p)
        bf[nt].u[p] = *(const unsigned*)&Bs[cur][col][kof + (p < 4 ? 2 * p : 16 + 2 * (p - 4))];
    }
    #pragma unroll
    for (int mt = 0; mt < 2; ++mt)
      #pragma unroll
      for (int nt = 0; nt < 4; ++nt)
        acc[mt][nt] = __builtin_amdgcn_wmma_f32_16x16x32_f16(
            false, af[mt].v, false, bf[nt].v, (short)0, acc[mt][nt], false, false);
    async_wait0();
    __syncthreads();
  }

  // Epilogue: C/D 16x16 f32 layout: row = v + 8*(lane>=16), col = lane%16
  #pragma unroll
  for (int mt = 0; mt < 2; ++mt) {
    #pragma unroll
    for (int nt = 0; nt < 4; ++nt) {
      #pragma unroll
      for (int v = 0; v < 8; ++v) {
        const int gm = m0 + waveM * 32 + mt * 16 + lhi * 8 + v;
        const int gn = n0 + waveN * 64 + nt * 16 + lmod;
        if (gm < M && gn < N) {
          float c = alpha * acc[mt][nt][v];
          if (bias) c += bias[gn];
          if (Res)  c += Res[(long long)gm * ldr + gn];
          if (Cf)   Cf[(long long)gm * ldcf + gn] = c;
          if (Ch)   Ch[(long long)gm * ldch + gn] = (half_t)c;
        }
      }
    }
  }
}

// ---------------------------------------------------------------------------
__global__ void cvt_f32_to_f16(const float* __restrict__ in, half_t* __restrict__ out,
                               long long n) {
  long long i = (long long)blockIdx.x * blockDim.x + threadIdx.x;
  const long long str = (long long)gridDim.x * blockDim.x;
  for (; i < n; i += str) out[i] = (half_t)in[i];
}

// cos-sim epilogue: one block (128 thr) per proposal row
__global__ __launch_bounds__(128)
void cos_out_kernel(const float* __restrict__ x, const float* __restrict__ q,
                    float* __restrict__ outS, float* __restrict__ outO,
                    int P, int Cc) {
  const int row = blockIdx.x;
  const int b   = row / P;
  const float* xr = x + (long long)row * Cc;
  const float* s  = q + (long long)b * 3 * Cc;
  const float* o  = s + Cc;
  float ds = 0.f, dq = 0.f, xx = 0.f, ss = 0.f, oo = 0.f;
  for (int c = threadIdx.x; c < Cc; c += blockDim.x) {
    const float xv = xr[c], sv = s[c], ov = o[c];
    ds += xv * sv; dq += xv * ov; xx += xv * xv; ss += sv * sv; oo += ov * ov;
  }
  __shared__ float red[5][128];
  red[0][threadIdx.x] = ds; red[1][threadIdx.x] = dq; red[2][threadIdx.x] = xx;
  red[3][threadIdx.x] = ss; red[4][threadIdx.x] = oo;
  __syncthreads();
  for (int off = 64; off > 0; off >>= 1) {
    if (threadIdx.x < off) {
      #pragma unroll
      for (int r = 0; r < 5; ++r) red[r][threadIdx.x] += red[r][threadIdx.x + off];
    }
    __syncthreads();
  }
  if (threadIdx.x == 0) {
    ds = red[0][0]; dq = red[1][0]; xx = red[2][0]; ss = red[3][0]; oo = red[4][0];
    float xs = ds / fmaxf(sqrtf(xx) * sqrtf(ss), 1e-8f);
    float xo = dq / fmaxf(sqrtf(xx) * sqrtf(oo), 1e-8f);
    xs = fminf((xs + 1.f) * 0.5f, 0.99f) + 1e-7f;
    xo = fminf((xo + 1.f) * 0.5f, 0.99f) + 1e-7f;
    outS[2 * row]     = 1.f - xs;
    outS[2 * row + 1] = xs;
    outO[2 * row]     = 1.f - xo;
    outO[2 * row + 1] = xo;
  }
}

// ---------------------------------------------------------------------------
static void gemm(hipStream_t s,
                 const half_t* A, long long sA, int lda, int colA,
                 const half_t* B, long long sB, int ldb, int bKN,
                 const float* bias,
                 const float* res, long long sRes, int ldr,
                 float alpha,
                 float* Cf, long long sCf, int ldcf,
                 half_t* Ch, long long sCh, int ldch,
                 int M, int N, int K, int batches)
{
  dim3 g((N + BN - 1) / BN, (M + BM - 1) / BM, batches);
  gemm_f16_wmma<<<g, dim3(256), 0, s>>>(A, sA, lda, colA, B, sB, ldb, bKN,
                                        bias, res, sRes, ldr, alpha,
                                        Cf, sCf, ldcf, Ch, sCh, ldch, M, N, K);
}

extern "C" void kernel_launch(void* const* d_in, const int* in_sizes, int n_in,
                              void* d_out, int out_size, void* d_ws, size_t ws_size,
                              hipStream_t stream) {
  (void)in_sizes; (void)n_in; (void)out_size; (void)ws_size;
  const int  CC = 1024, NDIM = 556, NBB = 604, BSz = 64, PR = 512;
  const long long MP = (long long)BSz * PR;            // 32768
  const float* F[33];
  for (int i = 0; i < 33; ++i) F[i] = (const float*)d_in[i];
  float* OUT = (float*)d_out;

  char* base = (char*)d_ws;
  size_t off = 0;
  auto allocB = [&](size_t bytes) -> void* {
    void* r = base + off;
    off += (bytes + 255) & ~(size_t)255;
    return r;
  };
  auto allocH = [&](long long elems) { return (half_t*)allocB((size_t)elems * 2); };
  auto allocF = [&](long long elems) { return (float*)allocB((size_t)elems * 4); };

  // f16 weights
  half_t* wh_sp  = allocH((long long)CC * NDIM);
  half_t* wh_op  = allocH((long long)CC * NDIM);
  half_t* wh_tr  = allocH((long long)CC * NDIM);
  half_t* wh_sag = allocH((long long)CC * CC);
  half_t* wh_saW = allocH((long long)CC * CC);
  half_t* wh_sat = allocH((long long)CC * CC);
  half_t* wh_sap = allocH((long long)CC * CC);
  half_t* wh_cag = allocH((long long)CC * CC);
  half_t* wh_cg2 = allocH((long long)CC * CC);
  half_t* wh_caW = allocH((long long)CC * CC);
  half_t* wh_caQ = allocH((long long)CC * CC);
  half_t* wh_cat = allocH((long long)CC * CC);
  half_t* wh_cap = allocH((long long)CC * CC);
  half_t* wh_bb  = allocH((long long)NBB * CC);
  // f16 activations
  half_t* sub_h = allocH((long long)BSz * NDIM);
  half_t* obj_h = allocH((long long)BSz * NDIM);
  half_t* rel_h = allocH((long long)BSz * NDIM);
  half_t* xh0   = allocH(MP * CC);   // f16 of original x_feat (x_box)
  half_t* xh    = allocH(MP * CC);   // working x (f16)
  half_t* t_h   = allocH(MP * CC);
  half_t* g_h   = allocH(MP * CC);
  half_t* p_h   = allocH(MP * CC);
  half_t* f_h   = allocH((long long)BSz * PR * PR);
  half_t* qh    = allocH((long long)BSz * 3 * CC);
  half_t* tq    = allocH((long long)BSz * 3 * CC);
  half_t* pq    = allocH((long long)BSz * 3 * CC);
  half_t* gq    = allocH((long long)BSz * 3 * CC);
  half_t* naq   = allocH((long long)BSz * 3 * CC);
  half_t* tq2   = allocH((long long)BSz * 3 * CC);
  half_t* g2q   = allocH((long long)BSz * 3 * CC);
  half_t* naq2  = allocH((long long)BSz * 3 * CC);
  half_t* fq    = allocH((long long)BSz * 9);
  half_t* fc    = allocH((long long)BSz * 6 * CC);
  // f32
  float* qf0  = allocF((long long)BSz * 3 * CC);
  float* qf1  = allocF((long long)BSz * 3 * CC);
  float* qf2  = allocF((long long)BSz * 3 * CC);
  float* xf1  = allocF(MP * CC);
  float* xfin = allocF(MP * CC);

  // -------- converts (fp32 -> fp16) --------
  auto cvt = [&](const float* src, half_t* dst, long long n) {
    int blocks = (int)((n + 255) / 256); if (blocks > 4096) blocks = 4096;
    cvt_f32_to_f16<<<blocks, 256, 0, stream>>>(src, dst, n);
  };
  cvt(F[5],  wh_sp,  (long long)CC * NDIM);
  cvt(F[7],  wh_op,  (long long)CC * NDIM);
  cvt(F[9],  wh_tr,  (long long)CC * NDIM);
  cvt(F[11], wh_sag, (long long)CC * CC);
  cvt(F[13], wh_saW, (long long)CC * CC);
  cvt(F[15], wh_sat, (long long)CC * CC);
  cvt(F[17], wh_sap, (long long)CC * CC);
  cvt(F[19], wh_cag, (long long)CC * CC);
  cvt(F[21], wh_cg2, (long long)CC * CC);
  cvt(F[23], wh_caW, (long long)CC * CC);
  cvt(F[25], wh_caQ, (long long)CC * CC);
  cvt(F[27], wh_cat, (long long)CC * CC);
  cvt(F[29], wh_cap, (long long)CC * CC);
  cvt(F[31], wh_bb,  (long long)NBB * CC);
  cvt(F[1],  sub_h,  (long long)BSz * NDIM);
  cvt(F[2],  obj_h,  (long long)BSz * NDIM);
  cvt(F[3],  rel_h,  (long long)BSz * NDIM);
  cvt(F[0],  xh0,    MP * CC);

  const long long SXB = (long long)PR * CC;   // 524288 per-batch x stride
  const long long SQB = 3LL * CC;             // 3072 per-batch q stride

  // -------- q assembly: s/o/r linears written interleaved (ldc = 3072) --------
  gemm(stream, sub_h, 0, NDIM, 0, wh_sp, 0, NDIM, 0, F[6],  nullptr, 0, 0, 1.f,
       qf0 + 0,    0, 3 * CC, qh + 0,    0, 3 * CC, BSz, CC, NDIM, 1);
  gemm(stream, obj_h, 0, NDIM, 0, wh_op, 0, NDIM, 0, F[8],  nullptr, 0, 0, 1.f,
       qf0 + CC,   0, 3 * CC, qh + CC,   0, 3 * CC, BSz, CC, NDIM, 1);
  gemm(stream, rel_h, 0, NDIM, 0, wh_tr, 0, NDIM, 0, F[10], nullptr, 0, 0, 1.f,
       qf0 + 2*CC, 0, 3 * CC, qh + 2*CC, 0, 3 * CC, BSz, CC, NDIM, 1);

  // -------- match_block on x --------
  gemm(stream, xh0, 0, CC, 0, wh_sat, 0, CC, 0, F[16], nullptr, 0, 0, 1.f,
       nullptr, 0, 0, t_h, 0, CC, (int)MP, CC, CC, 1);                        // theta lin
  gemm(stream, xh0, 0, CC, 0, wh_sap, 0, CC, 0, F[18], nullptr, 0, 0, 1.f,
       nullptr, 0, 0, p_h, 0, CC, (int)MP, CC, CC, 1);                        // phi lin
  gemm(stream, xh0, 0, CC, 0, wh_sag, 0, CC, 0, F[12], nullptr, 0, 0, 1.f,
       nullptr, 0, 0, g_h, 0, CC, (int)MP, CC, CC, 1);                        // g lin
  // f_div = (theta_x @ phi_x)/512 : A = t (col-major view, ld=512); B = p in (K,N) ld=512
  gemm(stream, t_h, SXB, PR, 1, p_h, SXB, PR, 1, nullptr, nullptr, 0, 0, 1.f / 512.f,
       nullptr, 0, 0, f_h, (long long)PR * PR, PR, PR, PR, CC, BSz);
  // non_aim (flattened) = d_x^T @ f_div^T : A = g row-major ld=512, B = f row-major ld=512
  gemm(stream, g_h, SXB, PR, 0, f_h, (long long)PR * PR, PR, 0, nullptr, nullptr, 0, 0, 1.f,
       nullptr, 0, 0, t_h, SXB, PR, CC, PR, PR, BSz);                         // reuse t_h
  // x1 = non_aim @ W^T + b + x
  gemm(stream, t_h, 0, CC, 0, wh_saW, 0, CC, 0, F[14], F[0], 0, CC, 1.f,
       xf1, 0, CC, xh, 0, CC, (int)MP, CC, CC, 1);

  // -------- match_block on q --------
  gemm(stream, qh, 0, CC, 0, wh_sat, 0, CC, 0, F[16], nullptr, 0, 0, 1.f,
       nullptr, 0, 0, tq, 0, CC, BSz * 3, CC, CC, 1);
  gemm(stream, qh, 0, CC, 0, wh_sap, 0, CC, 0, F[18], nullptr, 0, 0, 1.f,
       nullptr, 0, 0, pq, 0, CC, BSz * 3, CC, CC, 1);
  gemm(stream, qh, 0, CC, 0, wh_sag, 0, CC, 0, F[12], nullptr, 0, 0, 1.f,
       nullptr, 0, 0, gq, 0, CC, BSz * 3, CC, CC, 1);
  gemm(stream, tq, SQB, 3, 1, pq, SQB, 3, 1, nullptr, nullptr, 0, 0, 1.f / 3.f,
       nullptr, 0, 0, fq, 9, 3, 3, 3, CC, BSz);
  gemm(stream, gq, SQB, 3, 0, fq, 9, 3, 0, nullptr, nullptr, 0, 0, 1.f,
       nullptr, 0, 0, naq, SQB, 3, CC, 3, 3, BSz);
  gemm(stream, naq, 0, CC, 0, wh_saW, 0, CC, 0, F[14], qf0, 0, CC, 1.f,
       qf1, 0, CC, qh, 0, CC, BSz * 3, CC, CC, 1);

  // -------- cross_block --------
  gemm(stream, xh, 0, CC, 0, wh_cag, 0, CC, 0, F[20], nullptr, 0, 0, 1.f,
       nullptr, 0, 0, g_h, 0, CC, (int)MP, CC, CC, 1);                        // g(detect)
  gemm(stream, xh, 0, CC, 0, wh_cap, 0, CC, 0, F[30], nullptr, 0, 0, 1.f,
       nullptr, 0, 0, p_h, 0, CC, (int)MP, CC, CC, 1);                        // phi(detect)
  gemm(stream, qh, 0, CC, 0, wh_cg2, 0, CC, 0, F[22], nullptr, 0, 0, 1.f,
       nullptr, 0, 0, g2q, 0, CC, BSz * 3, CC, CC, 1);                        // g2(aim)
  gemm(stream, qh, 0, CC, 0, wh_cat, 0, CC, 0, F[28], nullptr, 0, 0, 1.f,
       nullptr, 0, 0, tq2, 0, CC, BSz * 3, CC, CC, 1);                        // theta(aim)
  // f = theta_x @ phi_x : M=6,N=1024,K=512 ; A col-major ld=6, B (K,N) ld=1024
  gemm(stream, tq2, SQB, 6, 1, p_h, SXB, CC, 1, nullptr, nullptr, 0, 0, 1.f,
       nullptr, 0, 0, fc, 6LL * CC, CC, 6, CC, PR, BSz);
  // non_aim flat = d_x^T @ (f/1024)^T : M=512,N=6,K=1024 ; A=g row-major ld=1024, B=f row-major
  gemm(stream, g_h, SXB, CC, 0, fc, 6LL * CC, CC, 0, nullptr, nullptr, 0, 0, 1.f / 1024.f,
       nullptr, 0, 0, naq2, SQB, 6, PR, 6, CC, BSz);
  // q2 = non_aim @ W^T + b + q1
  gemm(stream, naq2, 0, CC, 0, wh_caW, 0, CC, 0, F[24], qf1, 0, CC, 1.f,
       qf2, 0, CC, nullptr, 0, 0, BSz * 3, CC, CC, 1);
  // non_det flat = a_x^T @ (f/6) : M=512,N=1024,K=6 ; A=g2 row-major ld=6, B=f in (K,N) ld=1024
  gemm(stream, g2q, SQB, 6, 0, fc, 6LL * CC, CC, 1, nullptr, nullptr, 0, 0, 1.f / 6.f,
       nullptr, 0, 0, p_h, SXB, CC, PR, CC, 6, BSz);                          // reuse p_h
  // x_final = non_det @ Q^T + b + x1
  gemm(stream, p_h, 0, CC, 0, wh_caQ, 0, CC, 0, F[26], xf1, 0, CC, 1.f,
       xfin, 0, CC, nullptr, 0, 0, (int)MP, CC, CC, 1);

  // -------- heads --------
  // bbox_deltas = x_feat @ bb^T + bb_b  -> d_out[131072 ...]
  gemm(stream, xh0, 0, CC, 0, wh_bb, 0, CC, 0, F[32], nullptr, 0, 0, 1.f,
       OUT + 2 * 65536, 0, NBB, nullptr, 0, 0, (int)MP, NBB, CC, 1);
  // x_subj / x_obj
  cos_out_kernel<<<dim3((unsigned)MP), dim3(128), 0, stream>>>(
      xfin, qf2, OUT, OUT + 65536, PR, CC);
}
